// SceneGraphModule_76965813944542
// MI455X (gfx1250) — matrix operand order repaired and checked by reading, not compiled, not run on hardware
//
#include <hip/hip_runtime.h>

// ---------------------------------------------------------------------------
// Sizes (fixed by the reference)
// ---------------------------------------------------------------------------
#define BB   8
#define SS   256
#define VH   1024
#define DD   256
#define HH   8
#define HD   32
#define MM   (BB * SS)          // 2048 rows for all token-level GEMMs

typedef __attribute__((ext_vector_type(16))) _Float16     v16h;
typedef __attribute__((ext_vector_type(8)))  _Float16     v8h;
typedef __attribute__((ext_vector_type(8)))  float        v8f;
typedef __attribute__((ext_vector_type(4)))  unsigned int u32x4;
typedef __attribute__((ext_vector_type(4)))  int          i32x4;
typedef __attribute__((ext_vector_type(8)))  int          i32x8;

#if defined(__has_builtin)
#if __has_builtin(__builtin_amdgcn_tensor_load_to_lds)
#define USE_TDM 1
#endif
#endif
#ifndef USE_TDM
#define USE_TDM 0
#endif

__device__ __forceinline__ v8f v8f_zero() {
  v8f z = {0.f, 0.f, 0.f, 0.f, 0.f, 0.f, 0.f, 0.f};
  return z;
}

union HalfFrag {
  v16h  v;
  u32x4 q[2];
};

// A fragment (16x32 f16, MxK): lane = row (lane&15); K halves are
// {hi*8 .. +8} and {16+hi*8 .. +8}, hi = lane>>4.  (ISA 7.12.2)
__device__ __forceinline__ v16h frag_a_lds(const _Float16* tile, int stride) {
  const int lane = threadIdx.x & 31;
  const int r = lane & 15, hi = lane >> 4;
  HalfFrag u;
  u.q[0] = *reinterpret_cast<const u32x4*>(tile + r * stride + hi * 8);
  u.q[1] = *reinterpret_cast<const u32x4*>(tile + r * stride + 16 + hi * 8);
  return u.v;
}

// B fragment (32x16 f16, KxN) from a transposed [N][K] LDS tile:
// lane = column (lane&15); K halves are {hi*8 .. +8} and {16+hi*8 .. +8}.
__device__ __forceinline__ v16h frag_b_lds(const _Float16* tileT, int stride) {
  const int lane = threadIdx.x & 31;
  const int n = lane & 15, hi = lane >> 4;
  HalfFrag u;
  u.q[0] = *reinterpret_cast<const u32x4*>(tileT + n * stride + hi * 8);
  u.q[1] = *reinterpret_cast<const u32x4*>(tileT + n * stride + 16 + hi * 8);
  return u.v;
}

#define WMMA_F16(a, b, c) \
  __builtin_amdgcn_wmma_f32_16x16x32_f16(false, (a), false, (b), (short)0, (c), false, false)

// ---------------------------------------------------------------------------
// Tile staging: Tensor Data Mover (preferred) or vectorized b128 copy fallback
// ---------------------------------------------------------------------------
__device__ __forceinline__ unsigned lds_addr_of(const void* p) {
  return (unsigned)(unsigned long long)p;   // low 32 bits of flat shared addr
}

__device__ __forceinline__ void tdm_load_tile_f16(unsigned lds_off,
                                                  const _Float16* g,
                                                  int tile_cols, int tile_rows,
                                                  int stride_elems) {
#if USE_TDM
  unsigned long long ga = (unsigned long long)g;
  u32x4 g0;
  g0[0] = 1u;                                            // count=1, user D#
  g0[1] = lds_off;                                       // lds_addr
  g0[2] = (unsigned)ga;                                  // global_addr[31:0]
  g0[3] = (unsigned)((ga >> 32) & 0x01ffffffull) | (2u << 30);  // addr hi | type=2
  i32x8 g1;
  g1[0] = 1 << 16;                                       // data_size = 2 bytes
  g1[1] = (stride_elems & 0xffff) << 16;                 // tensor_dim0[15:0]
  g1[2] = ((unsigned)stride_elems >> 16) | ((tile_rows & 0xffff) << 16); // dim0 hi | dim1 lo
  g1[3] = (tile_cols & 0xffff) << 16;                    // dim1 hi=0 | tile_dim0
  g1[4] = tile_rows & 0xffff;                            // tile_dim1 | tile_dim2=0
  g1[5] = stride_elems;                                  // tensor_dim0_stride[31:0]
  g1[6] = 0;
  g1[7] = 0;
  i32x4 z4 = {0, 0, 0, 0};
#if defined(__clang_major__) && (__clang_major__ >= 23)
  i32x8 z8 = {0, 0, 0, 0, 0, 0, 0, 0};
  __builtin_amdgcn_tensor_load_to_lds(g0, g1, z4, z4, z8, 0);
#else
  __builtin_amdgcn_tensor_load_to_lds(g0, g1, z4, z4, 0);
#endif
#endif
}

// cols must be a multiple of 8; all 16B aligned
__device__ __forceinline__ void stage_tile_f16(_Float16* dst, int dstride,
                                               const _Float16* src, int sstride,
                                               int rows, int cols) {
  const int tid = threadIdx.x;
  const int cpr = cols >> 3;
  for (int i = tid; i < rows * cpr; i += 256) {
    const int r = i / cpr, c = (i - r * cpr) * 8;
    *reinterpret_cast<u32x4*>(dst + r * dstride + c) =
        *reinterpret_cast<const u32x4*>(src + r * sstride + c);
  }
}

// ---------------------------------------------------------------------------
// Prep: f32->f16 convert, and [K,N]f32 -> [N,K]f16 weight transpose
// ---------------------------------------------------------------------------
__global__ __launch_bounds__(256) void cvt_f16_kernel(const float* __restrict__ s,
                                                      _Float16* __restrict__ d, int n) {
  const int i = blockIdx.x * 256 + threadIdx.x;
  if (i < n) d[i] = (_Float16)s[i];
}

__global__ __launch_bounds__(256) void transpose_f16_kernel(const float* __restrict__ W,
                                                            _Float16* __restrict__ Wt,
                                                            int K, int N) {
  const int i = blockIdx.x * 256 + threadIdx.x;
  if (i < K * N) {
    const int k = i / N, n = i - k * N;
    Wt[n * K + k] = (_Float16)W[i];
  }
}

// ---------------------------------------------------------------------------
// Generic GEMM: out = act(A[M,K]f16 @ Wt[N,ldk]f16^T + bias) (+ resid)
// writes f32 C (optional) and/or f16 Ch (optional)
// block tile 64x64, 8 waves -> two 16x16 WMMA tiles each (shared A fragment)
// TDM double-buffered: next K-slab streams while current one is consumed.
// ---------------------------------------------------------------------------
#define GEMM_BM 64
#define GEMM_BN 64
#define GEMM_BK 32

__global__ __launch_bounds__(256) void gemm_kernel(
    const _Float16* __restrict__ A, const _Float16* __restrict__ Wt,
    const float* __restrict__ bias, const float* __restrict__ resid,
    float* __restrict__ C, _Float16* __restrict__ Ch,
    int M, int K, int N, int ldk, int relu) {
  __shared__ _Float16 sA[2][GEMM_BM][GEMM_BK];  // 8 KB
  __shared__ _Float16 sB[2][GEMM_BN][GEMM_BK];  // 8 KB (stored [n][k])
  const int tid  = threadIdx.x;
  const int wave = tid >> 5;
  const int lane = tid & 31;
  const int m0 = blockIdx.x * GEMM_BM;
  const int n0 = blockIdx.y * GEMM_BN;
  const int wm = (wave >> 1) * 16;   // 4 m-tiles
  const int wn = (wave & 1) * 32;    // 2 n-columns; wave covers wn and wn+16
  const int nsteps = K / GEMM_BK;

  v8f acc0 = v8f_zero();
  v8f acc1 = v8f_zero();

#if USE_TDM
  if (wave == 0) {
    tdm_load_tile_f16(lds_addr_of(&sA[0][0][0]), A + m0 * K, GEMM_BK, GEMM_BM, K);
    tdm_load_tile_f16(lds_addr_of(&sB[0][0][0]), Wt + n0 * ldk, GEMM_BK, GEMM_BN, ldk);
  }
#endif
  for (int s = 0; s < nsteps; ++s) {
    const int cur = s & 1;
    const int kb = s * GEMM_BK;
#if USE_TDM
    if (wave == 0) {
      if (s + 1 < nsteps) {
        const int nxt = cur ^ 1;
        tdm_load_tile_f16(lds_addr_of(&sA[nxt][0][0]),
                          A + m0 * K + kb + GEMM_BK, GEMM_BK, GEMM_BM, K);
        tdm_load_tile_f16(lds_addr_of(&sB[nxt][0][0]),
                          Wt + n0 * ldk + kb + GEMM_BK, GEMM_BK, GEMM_BN, ldk);
        __builtin_amdgcn_s_wait_tensorcnt(2);  // in-order: current slab landed
      } else {
        __builtin_amdgcn_s_wait_tensorcnt(0);
      }
    }
    __syncthreads();  // current buffers ready for all waves
#else
    __syncthreads();
    stage_tile_f16(&sA[cur][0][0], GEMM_BK, A + m0 * K + kb, K, GEMM_BM, GEMM_BK);
    stage_tile_f16(&sB[cur][0][0], GEMM_BK, Wt + n0 * ldk + kb, ldk, GEMM_BN, GEMM_BK);
    __syncthreads();
#endif
    v16h af  = frag_a_lds(&sA[cur][wm][0], GEMM_BK);
    v16h bf0 = frag_b_lds(&sB[cur][wn][0], GEMM_BK);
    v16h bf1 = frag_b_lds(&sB[cur][wn + 16][0], GEMM_BK);
    acc0 = WMMA_F16(af, bf0, acc0);
    acc1 = WMMA_F16(af, bf1, acc1);
#if USE_TDM
    __syncthreads();  // fragments consumed before this buffer is re-filled
#endif
  }

  const int mbase = m0 + wm + ((lane >> 4) << 3);
  const int nc0   = n0 + wn + (lane & 15);
  const int nc1   = nc0 + 16;
  const float bv0 = bias ? bias[nc0] : 0.f;
  const float bv1 = bias ? bias[nc1] : 0.f;
  for (int r = 0; r < 8; ++r) {
    const int row = mbase + r;
    float v0 = acc0[r] + bv0;
    float v1 = acc1[r] + bv1;
    if (relu) { v0 = fmaxf(v0, 0.f); v1 = fmaxf(v1, 0.f); }
    if (resid) { v0 += resid[row * N + nc0]; v1 += resid[row * N + nc1]; }
    if (C) { C[row * N + nc0] = v0; C[row * N + nc1] = v1; }
    if (Ch) { Ch[row * N + nc0] = (_Float16)v0; Ch[row * N + nc1] = (_Float16)v1; }
  }
}

// ---------------------------------------------------------------------------
// coords layer 2: out[2048,3] = h[2048,128] @ w[128,3] + b   (tiny, VALU)
// ---------------------------------------------------------------------------
__global__ __launch_bounds__(256) void coord2_kernel(
    const float* __restrict__ h, const float* __restrict__ w,
    const float* __restrict__ b, float* __restrict__ out) {
  const int idx = blockIdx.x * 256 + threadIdx.x;
  if (idx >= MM * 3) return;
  const int row = idx / 3, n = idx - row * 3;
  float acc = b[n];
  for (int k = 0; k < 128; ++k) acc += h[row * 128 + k] * w[k * 3 + n];
  out[idx] = acc;
}

// ---------------------------------------------------------------------------
// Fused relation MLP. Block = batch b, 16 i-rows x 4 j-rows = 64 pairs.
// H=relu(a_i+b_j+bias1)[64,256] -> relu(H@W2+b2)[64,128] -> @W3+b3 [64,64]
// grid = BB*(SS/16)*(SS/4) = 8192 blocks; W2t/W3t are pre-transposed f16.
// ---------------------------------------------------------------------------
__global__ __launch_bounds__(256) void rel_kernel(
    const _Float16* __restrict__ Arel, const _Float16* __restrict__ Brel,
    const _Float16* __restrict__ b1h, const _Float16* __restrict__ W2t,
    const float* __restrict__ b2, const _Float16* __restrict__ W3t,
    const float* __restrict__ b3, float* __restrict__ out) {
  __shared__ _Float16 sH[64][256];    // 32 KB (later reused for H2[64][128])
  __shared__ _Float16 sW2t[128][32];  // 8 KB, W2^T K-slab
  __shared__ _Float16 sW3t[64][128];  // 16 KB, full W3^T
  const int tid  = threadIdx.x;
  const int wave = tid >> 5;
  const int lane = tid & 31;
  const int blk = blockIdx.x;
  const int j0 = (blk & 63) * 4;
  const int i0 = ((blk >> 6) & 15) * 16;
  const int b  = blk >> 10;
  const int arow = b * SS + i0;
  const int brow = b * SS + j0;

  // W3^T [64][128] is contiguous -> one DMA (or vector copy)
#if USE_TDM
  if (wave == 0)
    tdm_load_tile_f16(lds_addr_of(&sW3t[0][0]), W3t, 128, 64, 128);
#else
  stage_tile_f16(&sW3t[0][0], 128, W3t, 128, 64, 128);
#endif

  // build H with packed f16 math: p = jj*16 + ii
  const _Float16 hzero = (_Float16)0.f;
  for (int i = tid; i < (64 * 256) / 8; i += 256) {
    const int idx = i * 8;
    const int p = idx >> 8, c = idx & 255;
    const int ii = p & 15, jj = p >> 4;
    v8h av = *reinterpret_cast<const v8h*>(Arel + (arow + ii) * DD + c);
    v8h bv = *reinterpret_cast<const v8h*>(Brel + (brow + jj) * DD + c);
    v8h kv = *reinterpret_cast<const v8h*>(b1h + c);
    v8h hv = av + bv + kv;
    v8h z = {hzero, hzero, hzero, hzero, hzero, hzero, hzero, hzero};
    hv = __builtin_elementwise_max(hv, z);
    *reinterpret_cast<v8h*>(&sH[p][c]) = hv;
  }

  // layer 2: M=64 (4 m-tiles), K=256, N=128 (8 n-tiles)
  const int mt = wave >> 1;
  const int nh = wave & 1;
  v8f acc2[4];
  for (int t = 0; t < 4; ++t) acc2[t] = v8f_zero();
  for (int kb = 0; kb < 256; kb += 32) {
    __syncthreads();
#if USE_TDM
    if (wave == 0) {
      tdm_load_tile_f16(lds_addr_of(&sW2t[0][0]), W2t + kb, 32, 128, 256);
      __builtin_amdgcn_s_wait_tensorcnt(0);
    }
#else
    stage_tile_f16(&sW2t[0][0], 32, W2t + kb, 256, 128, 32);
#endif
    __syncthreads();
    v16h hf = frag_a_lds(&sH[mt * 16][kb], 256);
    for (int t = 0; t < 4; ++t) {
      v16h wf = frag_b_lds(&sW2t[(nh * 4 + t) * 16][0], 32);
      acc2[t] = WMMA_F16(hf, wf, acc2[t]);
    }
  }
  __syncthreads();

  // H2 = relu(acc2 + b2) into the (now dead) sH region as [64][128] f16
  _Float16* H2 = &sH[0][0];
  {
    const int mb = mt * 16 + ((lane >> 4) << 3);
    const int nn = lane & 15;
    for (int t = 0; t < 4; ++t) {
      const int n = (nh * 4 + t) * 16 + nn;
      const float bb = b2[n];
      for (int r = 0; r < 8; ++r) {
        float v = acc2[t][r] + bb;
        H2[(mb + r) * 128 + n] = (_Float16)fmaxf(v, 0.f);
      }
    }
  }
  __syncthreads();

  // layer 3: M=64, K=128, N=64 (4 n-tiles)
  v8f acc3[2];
  acc3[0] = v8f_zero();
  acc3[1] = v8f_zero();
  for (int kb = 0; kb < 128; kb += 32) {
    v16h hf = frag_a_lds(H2 + (mt * 16) * 128 + kb, 128);
    for (int t = 0; t < 2; ++t) {
      v16h wf = frag_b_lds(&sW3t[(nh * 2 + t) * 16][kb], 128);
      acc3[t] = WMMA_F16(hf, wf, acc3[t]);
    }
  }
  {
    const int mb = mt * 16 + ((lane >> 4) << 3);
    const int nn = lane & 15;
    for (int t = 0; t < 2; ++t) {
      const int n = (nh * 2 + t) * 16 + nn;
      const float bb = b3[n];
      for (int r = 0; r < 8; ++r) {
        const int p = mb + r;
        const int i = i0 + (p & 15);
        const int j = j0 + (p >> 4);
        out[(((long)b * SS + i) * SS + j) * 64 + n] = acc3[t][r] + bb;
      }
    }
  }
}

// ---------------------------------------------------------------------------
// Fused attention core for one (b, h, 16-query tile), f16 in / f16 ctx out
// grid = BB * HH * (SS/16) = 1024 blocks
// ---------------------------------------------------------------------------
__global__ __launch_bounds__(256) void attn_kernel(
    const _Float16* __restrict__ Q, const _Float16* __restrict__ Km,
    const _Float16* __restrict__ V, _Float16* __restrict__ ctx) {
  __shared__ _Float16 sK[SS][HD];   // 16 KB  [key s][hd]  (B^T for QK^T)
  __shared__ _Float16 sVt[HD][SS];  // 16 KB  V^T [hd][key s] (B^T for P*V)
  __shared__ _Float16 sQ[16][HD];   // 1 KB
  __shared__ float    sS[16][SS];   // 16 KB scores
  __shared__ _Float16 sP[16][SS];   // 8 KB probabilities
  const int tid  = threadIdx.x;
  const int wave = tid >> 5;
  const int lane = tid & 31;
  const int blk = blockIdx.x;
  const int q0 = (blk & 15) * 16;
  const int h  = (blk >> 4) & 7;
  const int b  = blk >> 7;
  const int rowbase = b * SS;
  const int colbase = h * HD;

#if USE_TDM
  if (wave == 0) {
    tdm_load_tile_f16(lds_addr_of(&sK[0][0]), Km + rowbase * DD + colbase, HD, SS, DD);
    tdm_load_tile_f16(lds_addr_of(&sQ[0][0]), Q + (rowbase + q0) * DD + colbase, HD, 16, DD);
    __builtin_amdgcn_s_wait_tensorcnt(0);
  }
#else
  stage_tile_f16(&sK[0][0], HD, Km + rowbase * DD + colbase, DD, SS, HD);
  stage_tile_f16(&sQ[0][0], HD, Q + (rowbase + q0) * DD + colbase, DD, 16, HD);
#endif
  // V^T staging (transpose): vector loads, scalar b16 stores
  for (int i = tid; i < (SS * HD) / 8; i += 256) {
    const int idx = i * 8;
    const int s = idx >> 5, d = idx & 31;
    v8h vv = *reinterpret_cast<const v8h*>(V + (rowbase + s) * DD + colbase + d);
    for (int e = 0; e < 8; ++e) sVt[d + e][s] = vv[e];
  }
  __syncthreads();

  // scores: 16 column tiles of 16x16, one WMMA each; 2 per wave
  const float scale = 0.17677669529663687f;  // 1/sqrt(32)
  v16h qf = frag_a_lds(&sQ[0][0], HD);
  for (int t = 0; t < 2; ++t) {
    const int nt = wave * 2 + t;
    v16h kf = frag_b_lds(&sK[nt * 16][0], HD);
    v8f acc = v8f_zero();
    acc = WMMA_F16(qf, kf, acc);
    const int mb = (lane >> 4) << 3;
    const int n  = nt * 16 + (lane & 15);
    for (int r = 0; r < 8; ++r) sS[mb + r][n] = acc[r] * scale;
  }
  __syncthreads();

  // row softmax
  if (tid < 16) {
    float mx = -1e30f;
    for (int j = 0; j < SS; ++j) mx = fmaxf(mx, sS[tid][j]);
    float sum = 0.f;
    for (int j = 0; j < SS; ++j) {
      float e = __expf(sS[tid][j] - mx);
      sS[tid][j] = e;
      sum += e;
    }
    const float inv = 1.f / sum;
    for (int j = 0; j < SS; ++j) sP[tid][j] = (_Float16)(sS[tid][j] * inv);
  }
  __syncthreads();

  // ctx tile = P[16,256] @ V[256,32]: 2 n-tiles, waves 0..1 (uniform branch)
  if (wave < 2) {
    v8f acc = v8f_zero();
    for (int kb = 0; kb < SS; kb += 32) {
      v16h pf = frag_a_lds(&sP[0][kb], SS);
      v16h vf = frag_b_lds(&sVt[wave * 16][kb], SS);
      acc = WMMA_F16(pf, vf, acc);
    }
    const int mb = (lane >> 4) << 3;
    const int n  = colbase + wave * 16 + (lane & 15);
    for (int r = 0; r < 8; ++r)
      ctx[(rowbase + q0 + mb + r) * DD + n] = (_Float16)acc[r];
  }
}

// ---------------------------------------------------------------------------
// Launch
// ---------------------------------------------------------------------------
extern "C" void kernel_launch(void* const* d_in, const int* in_sizes, int n_in,
                              void* d_out, int out_size, void* d_ws, size_t ws_size,
                              hipStream_t stream) {
  (void)in_sizes; (void)n_in; (void)out_size; (void)ws_size;
  const float* features = (const float*)d_in[0];
  const float* det_w1   = (const float*)d_in[1];
  const float* det_b1   = (const float*)d_in[2];
  const float* det_w2   = (const float*)d_in[3];
  const float* det_b2   = (const float*)d_in[4];
  const float* coord_w1 = (const float*)d_in[5];
  const float* coord_b1 = (const float*)d_in[6];
  const float* coord_w2 = (const float*)d_in[7];
  const float* coord_b2 = (const float*)d_in[8];
  const float* rel_w1   = (const float*)d_in[9];
  const float* rel_b1   = (const float*)d_in[10];
  const float* rel_w2   = (const float*)d_in[11];
  const float* rel_b2   = (const float*)d_in[12];
  const float* rel_w3   = (const float*)d_in[13];
  const float* rel_b3   = (const float*)d_in[14];
  const float* wq = (const float*)d_in[15];
  const float* bq = (const float*)d_in[16];
  const float* wk = (const float*)d_in[17];
  const float* bk = (const float*)d_in[18];
  const float* wv = (const float*)d_in[19];
  const float* bv = (const float*)d_in[20];
  const float* wo = (const float*)d_in[21];
  const float* bo = (const float*)d_in[22];

  float* out     = (float*)d_out;
  float* refined = out;                      // [2048,256] f32
  float* coords  = out + MM * DD;            // [2048,3]
  float* rel     = out + MM * DD + MM * 3;   // [8,256,256,64]

  // workspace carving (256B aligned)
  char* wsp = (char*)d_ws;
  auto alloc = [&](size_t bytes) -> void* {
    void* r = (void*)wsp;
    wsp += (bytes + 255) & ~(size_t)255;
    return r;
  };
  _Float16* features_h = (_Float16*)alloc((size_t)MM * VH * 2);
  _Float16* det_w1t    = (_Float16*)alloc((size_t)512 * VH * 2);
  _Float16* det_w2t    = (_Float16*)alloc((size_t)DD * 512 * 2);
  _Float16* coord_w1t  = (_Float16*)alloc((size_t)128 * DD * 2);
  _Float16* rel_w1t    = (_Float16*)alloc((size_t)DD * 512 * 2);  // [256][512]
  _Float16* rel_w2t    = (_Float16*)alloc((size_t)128 * DD * 2);
  _Float16* rel_w3t    = (_Float16*)alloc((size_t)64 * 128 * 2);
  _Float16* rel_b1h    = (_Float16*)alloc((size_t)DD * 2);
  _Float16* wqt        = (_Float16*)alloc((size_t)3 * DD * DD * 2);
  _Float16* wkt        = (_Float16*)alloc((size_t)3 * DD * DD * 2);
  _Float16* wvt        = (_Float16*)alloc((size_t)3 * DD * DD * 2);
  _Float16* wot        = (_Float16*)alloc((size_t)3 * DD * DD * 2);
  _Float16* det_h_h    = (_Float16*)alloc((size_t)MM * 512 * 2);
  float*    obj        = (float*)alloc((size_t)MM * DD * 4);
  _Float16* obj_h      = (_Float16*)alloc((size_t)MM * DD * 2);
  float*    coordh     = (float*)alloc((size_t)MM * 128 * 4);
  _Float16* a_h        = (_Float16*)alloc((size_t)MM * DD * 2);
  _Float16* b_h        = (_Float16*)alloc((size_t)MM * DD * 2);
  _Float16* qh         = (_Float16*)alloc((size_t)MM * DD * 2);
  _Float16* kh         = (_Float16*)alloc((size_t)MM * DD * 2);
  _Float16* vh         = (_Float16*)alloc((size_t)MM * DD * 2);
  _Float16* ctx_h      = (_Float16*)alloc((size_t)MM * DD * 2);
  _Float16* refined_h  = (_Float16*)alloc((size_t)MM * DD * 2);

  const dim3 blk(256);
  auto grid1 = [](int n) { return dim3((n + 255) / 256); };

  // ---- prep: f16 conversion + weight transposes ----
  cvt_f16_kernel<<<grid1(MM * VH), blk, 0, stream>>>(features, features_h, MM * VH);
  cvt_f16_kernel<<<grid1(DD), blk, 0, stream>>>(rel_b1, rel_b1h, DD);
  transpose_f16_kernel<<<grid1(VH * 512), blk, 0, stream>>>(det_w1, det_w1t, VH, 512);
  transpose_f16_kernel<<<grid1(512 * DD), blk, 0, stream>>>(det_w2, det_w2t, 512, DD);
  transpose_f16_kernel<<<grid1(DD * 128), blk, 0, stream>>>(coord_w1, coord_w1t, DD, 128);
  transpose_f16_kernel<<<grid1(512 * DD), blk, 0, stream>>>(rel_w1, rel_w1t, 512, DD);
  transpose_f16_kernel<<<grid1(DD * 128), blk, 0, stream>>>(rel_w2, rel_w2t, DD, 128);
  transpose_f16_kernel<<<grid1(128 * 64), blk, 0, stream>>>(rel_w3, rel_w3t, 128, 64);
  for (int l = 0; l < 3; ++l) {
    transpose_f16_kernel<<<grid1(DD * DD), blk, 0, stream>>>(wq + l * DD * DD, wqt + l * DD * DD, DD, DD);
    transpose_f16_kernel<<<grid1(DD * DD), blk, 0, stream>>>(wk + l * DD * DD, wkt + l * DD * DD, DD, DD);
    transpose_f16_kernel<<<grid1(DD * DD), blk, 0, stream>>>(wv + l * DD * DD, wvt + l * DD * DD, DD, DD);
    transpose_f16_kernel<<<grid1(DD * DD), blk, 0, stream>>>(wo + l * DD * DD, wot + l * DD * DD, DD, DD);
  }

  // ---- object detector MLP ----
  gemm_kernel<<<dim3(MM / 64, 512 / 64), blk, 0, stream>>>(
      features_h, det_w1t, det_b1, nullptr, nullptr, det_h_h, MM, VH, 512, VH, 1);
  gemm_kernel<<<dim3(MM / 64, DD / 64), blk, 0, stream>>>(
      det_h_h, det_w2t, det_b2, nullptr, obj, obj_h, MM, 512, DD, 512, 0);
  hipMemcpyAsync(refined, obj, (size_t)MM * DD * sizeof(float),
                 hipMemcpyDeviceToDevice, stream);
  hipMemcpyAsync(refined_h, obj_h, (size_t)MM * DD * sizeof(_Float16),
                 hipMemcpyDeviceToDevice, stream);

  // ---- coordinate predictor ----
  gemm_kernel<<<dim3(MM / 64, 128 / 64), blk, 0, stream>>>(
      obj_h, coord_w1t, coord_b1, nullptr, coordh, nullptr, MM, DD, 128, DD, 1);
  coord2_kernel<<<grid1(MM * 3), blk, 0, stream>>>(coordh, coord_w2, coord_b2, coords);

  // ---- relation predictor ----
  gemm_kernel<<<dim3(MM / 64, DD / 64), blk, 0, stream>>>(
      obj_h, rel_w1t, nullptr, nullptr, nullptr, a_h, MM, DD, DD, 512, 0);
  gemm_kernel<<<dim3(MM / 64, DD / 64), blk, 0, stream>>>(
      obj_h, rel_w1t + DD, nullptr, nullptr, nullptr, b_h, MM, DD, DD, 512, 0);
  rel_kernel<<<BB * (SS / 16) * (SS / 4), blk, 0, stream>>>(
      a_h, b_h, rel_b1h, rel_w2t, rel_b2, rel_w3t, rel_b3, rel);

  // ---- 3 self-attention layers (no residual add) ----
  for (int l = 0; l < 3; ++l) {
    const _Float16* wqtl = wqt + l * DD * DD; const float* bql = bq + l * DD;
    const _Float16* wktl = wkt + l * DD * DD; const float* bkl = bk + l * DD;
    const _Float16* wvtl = wvt + l * DD * DD; const float* bvl = bv + l * DD;
    const _Float16* wotl = wot + l * DD * DD; const float* bol = bo + l * DD;
    gemm_kernel<<<dim3(MM / 64, DD / 64), blk, 0, stream>>>(
        refined_h, wqtl, bql, nullptr, nullptr, qh, MM, DD, DD, DD, 0);
    gemm_kernel<<<dim3(MM / 64, DD / 64), blk, 0, stream>>>(
        refined_h, wktl, bkl, nullptr, nullptr, kh, MM, DD, DD, DD, 0);
    gemm_kernel<<<dim3(MM / 64, DD / 64), blk, 0, stream>>>(
        refined_h, wvtl, bvl, nullptr, nullptr, vh, MM, DD, DD, DD, 0);
    attn_kernel<<<BB * HH * (SS / 16), blk, 0, stream>>>(qh, kh, vh, ctx_h);
    gemm_kernel<<<dim3(MM / 64, DD / 64), blk, 0, stream>>>(
        ctx_h, wotl, bol, nullptr, refined, refined_h, MM, DD, DD, DD, 0);
  }
}
